// HW_att_12893491822789
// MI455X (gfx1250) — matrix operand order repaired and checked by reading, not compiled, not run on hardware
//
#include <hip/hip_runtime.h>

typedef __attribute__((ext_vector_type(16))) _Float16 v16h;
typedef __attribute__((ext_vector_type(8)))  float    v8f;

#define N_BATCH 32
#define C_CH    32
#define HW      2304            // 48*48
#define CHW     73728           // 32*2304
#define LOG2E   1.44269504088896340736f

// ---------------------------------------------------------------------------
// Phase 1: x_pool = mean_c + max_c ; q = Wq*pool+bq ; k = Wk*pool+bk ;
// per-batch kmax/kmin (analytic softmax row-max, since energy is rank-1).
// ---------------------------------------------------------------------------
__global__ void __launch_bounds__(256)
pool_qk_kernel(const float* __restrict__ x,
               const float* __restrict__ Wq, const float* __restrict__ bq,
               const float* __restrict__ Wk, const float* __restrict__ bk,
               float* __restrict__ qbuf, float* __restrict__ kbuf,
               float* __restrict__ kmm) {
  const int n   = blockIdx.x;
  const int tid = threadIdx.x;
  const float wq = Wq[0], bq0 = bq[0], wk = Wk[0], bk0 = bk[0];
  const float* xb = x + (size_t)n * CHW;

  float kmx = -3.4e38f, kmn = 3.4e38f;
  for (int p = tid; p < HW; p += 256) {
    float s = 0.f, mx = -3.4e38f;
#pragma unroll
    for (int c = 0; c < C_CH; ++c) {
      float v = xb[c * HW + p];        // coalesced across lanes
      s += v;
      mx = fmaxf(mx, v);
    }
    float pool = s * (1.0f / C_CH) + mx;
    float q = fmaf(wq, pool, bq0);
    float k = fmaf(wk, pool, bk0);
    qbuf[n * HW + p] = q;
    kbuf[n * HW + p] = k;
    kmx = fmaxf(kmx, k);
    kmn = fminf(kmn, k);
  }

  __shared__ float smx[256], smn[256];
  smx[tid] = kmx; smn[tid] = kmn;
  __syncthreads();
  for (int off = 128; off > 0; off >>= 1) {
    if (tid < off) {
      smx[tid] = fmaxf(smx[tid], smx[tid + off]);
      smn[tid] = fminf(smn[tid], smn[tid + off]);
    }
    __syncthreads();
  }
  if (tid == 0) { kmm[n * 2] = smx[0]; kmm[n * 2 + 1] = smn[0]; }
}

// ---------------------------------------------------------------------------
// Phase 2: v_flat[n][o][p] = sum_c Wv[o,c]*x[n,c,p] + bv[o], stored as f16.
// The torch .view() makes GEMM B-tile rows contiguous: v_r[j,d]=v_flat[j*32+d],
// so each 32-half row is one aligned 64-byte run -> phase 3 reads B as single halves v_r[j0+K, r].
// ---------------------------------------------------------------------------
__global__ void __launch_bounds__(256)
vproj_kernel(const float* __restrict__ x,
             const float* __restrict__ Wv, const float* __restrict__ bv,
             _Float16* __restrict__ vbuf_h) {
  const int idx = blockIdx.x * 256 + threadIdx.x;   // exact grid: n*o*p
  const int n   = idx / CHW;
  const int rem = idx - n * CHW;
  const int o   = rem / HW;
  const int p   = rem - o * HW;
  const float* xb = x + (size_t)n * CHW + p;
  float acc = bv[o];
#pragma unroll
  for (int c = 0; c < C_CH; ++c)
    acc = fmaf(Wv[o * C_CH + c], xb[c * HW], acc);  // x re-reads hit L2 (192MB)
  vbuf_h[idx] = (_Float16)acc;
}

// ---------------------------------------------------------------------------
// Phase 3: flash-style  out[i,:] = (1/Z_i) * sum_j exp(q_i*k_j - m_i)*v_r[j,:]
// 256-thread blocks (8 waves); k row (9KB) staged into LDS ONCE per block via
// async global->LDS DMA (global_load_async_to_lds_b128 + s_wait_asynccnt).
// Each wave owns one 16-row block: per K=32 step builds the exp tile as WMMA-A
// (f16, ISA lane layout; one v_fma + v_exp per element via log2e pre-scale)
// and issues THREE v_wmma_f32_16x16x32_f16: two for the N=32 output and one
// vs. all-ones B that accumulates the softmax denominators Z_M directly into
// the lanes that need them (D layout: M = v + 8*half).
// ---------------------------------------------------------------------------
__global__ void __launch_bounds__(256)
attn_wmma_kernel(const float* __restrict__ x,
                 const float* __restrict__ qbuf, const float* __restrict__ kbuf,
                 const float* __restrict__ kmm,  const _Float16* __restrict__ vbuf_h,
                 const float* __restrict__ gamma,
                 float* __restrict__ out) {
  const int n    = blockIdx.y;          // 0..31 batch
  const int tid  = threadIdx.x;
  const int wave = tid >> 5;            // 0..7
  const int lane = tid & 31;            // wave32
  const int half = lane >> 4;
  const int r    = lane & 15;
  const int i0   = (blockIdx.x * 8 + wave) * 16;   // row block of this wave

  __shared__ __align__(16) float k_lds[HW];        // 9 KB, staged once

  // ---- async DMA: kbuf[n*HW .. +2304) -> k_lds, 16B per lane per issue ----
  {
    const float* kbase = kbuf + (size_t)n * HW;    // uniform (SGPR pair)
    const unsigned lds_base = (unsigned)(unsigned long long)(&k_lds[0]);
    for (int t = tid; t < HW / 4; t += 256) {      // 576 x float4 chunks
      unsigned ldsoff = lds_base + (unsigned)t * 16u;
      unsigned goff   = (unsigned)t * 16u;
      asm volatile("global_load_async_to_lds_b128 %0, %1, %2"
                   :: "v"(ldsoff), "v"(goff), "s"(kbase) : "memory");
    }
    asm volatile("s_wait_asynccnt 0x0" ::: "memory");
  }
  __syncthreads();

  const float kmax = kmm[n * 2], kmin = kmm[n * 2 + 1];
  const float qr = qbuf[(size_t)n * HW + i0 + r];
  const float mr = (qr >= 0.f) ? qr * kmax : qr * kmin;  // exact row max (rank-1)
  // pre-scale by log2(e): e^(q*k-m) = 2^(qs*k + msn)  -> one v_fma + v_exp each
  const float qs  = qr * LOG2E;
  const float msn = -mr * LOG2E;

  v16h ONES;
#pragma unroll
  for (int t = 0; t < 16; ++t) ONES[t] = (_Float16)1.0f;

  v8f acc0 = {};   // N = 0..15
  v8f acc1 = {};   // N = 16..31
  v8f accz = {};   // Z accumulator: D[M,N] = sum_j e[M,j]  (all N identical)

  const _Float16* vb = vbuf_h + (size_t)n * CHW;

  for (int j0 = 0; j0 < HW; j0 += 32) {
    // A tile e[r,K], 16-bit A layout:
    //  lanes 0-15 : elems 0..7 -> K 0..7,  elems 8..15 -> K 16..23
    //  lanes 16-31: elems 0..7 -> K 8..15, elems 8..15 -> K 24..31
    v16h A;
#pragma unroll
    for (int t = 0; t < 16; ++t) {
      const int K = (t < 8 ? t : t + 8) + (half << 3);
      A[t] = (_Float16)__builtin_amdgcn_exp2f(fmaf(qs, k_lds[j0 + K], msn));
    }

    // B tiles: B[K][N] = v_r[j0+K, N]; lane holds column N=r (B1: 16+r), elems over K in A's K order
    const v16h* vrow = (const v16h*)(vb + (size_t)(j0 + lane) * 32);
    __builtin_prefetch((const void*)(vrow + 64), 0, 1);  // next j-step tile
    v16h B0, B1;
#pragma unroll
    for (int t = 0; t < 16; ++t) {
      const int K = (t < 8 ? t : t + 8) + (half << 3);
      B0[t] = vb[(size_t)(j0 + K) * 32 + r];
      B1[t] = vb[(size_t)(j0 + K) * 32 + 16 + r];
    }

    acc0 = __builtin_amdgcn_wmma_f32_16x16x32_f16(
        false, A, false, B0, (short)0, acc0, false, false);
    acc1 = __builtin_amdgcn_wmma_f32_16x16x32_f16(
        false, A, false, B1, (short)0, acc1, false, false);
    accz = __builtin_amdgcn_wmma_f32_16x16x32_f16(
        false, A, false, ONES, (short)0, accz, false, false);
  }

  // accz element v at this lane == Z for M=v+8*half — exactly our rows.
  const float gm = gamma[0];
  const size_t nbase = (size_t)n * CHW;
#pragma unroll
  for (int v = 0; v < 8; ++v) {
    const int M = v + (half << 3);
    const float invZ = __builtin_amdgcn_rcpf(accz[v]);   // v_rcp_f32, no div expansion
    const size_t base = nbase + (size_t)(i0 + M) * 32;
    out[base + r]      = fmaf(gm, acc0[v] * invZ, x[base + r]);
    out[base + 16 + r] = fmaf(gm, acc1[v] * invZ, x[base + 16 + r]);
  }
}

// ---------------------------------------------------------------------------
extern "C" void kernel_launch(void* const* d_in, const int* in_sizes, int n_in,
                              void* d_out, int out_size, void* d_ws, size_t ws_size,
                              hipStream_t stream) {
  const float* x     = (const float*)d_in[0];
  const float* Wq    = (const float*)d_in[1];
  const float* bq    = (const float*)d_in[2];
  const float* Wk    = (const float*)d_in[3];
  const float* bk    = (const float*)d_in[4];
  const float* Wv    = (const float*)d_in[5];
  const float* bv    = (const float*)d_in[6];
  const float* gamma = (const float*)d_in[7];
  float* out = (float*)d_out;

  float* ws   = (float*)d_ws;
  float* qbuf = ws;                              // 73728 f32
  float* kbuf = ws + 73728;                      // 73728 f32
  float* kmm  = ws + 147456;                     // 64 f32
  _Float16* vbuf_h = (_Float16*)(ws + 147520);   // 2359296 f16 (~4.7 MB)

  pool_qk_kernel<<<N_BATCH, 256, 0, stream>>>(x, Wq, bq, Wk, bk, qbuf, kbuf, kmm);

  vproj_kernel<<<(N_BATCH * C_CH * HW) / 256, 256, 0, stream>>>(x, Wv, bv, vbuf_h);

  dim3 g3(HW / 16 / 8, N_BATCH);   // 18 x 32 blocks, 8 waves each (1 row-block/wave)
  attn_wmma_kernel<<<g3, 256, 0, stream>>>(x, qbuf, kbuf, kmm, vbuf_h, gamma, out);
}
